// SelfAttention_32469952758093
// MI455X (gfx1250) — compile-verified, matrix operand order checked by reading, not run on hardware
//
#include <hip/hip_runtime.h>
#include <hip/hip_bf16.h>

// ---------------------------------------------------------------------------
// Problem constants (from reference): B=4, D=1024, L=2048, H=16, dh=64
// ---------------------------------------------------------------------------
#define BB    4
#define DD    1024
#define LL    2048
#define HH    16
#define DH    64
#define NEG_INF (-1.0e30f)

typedef __attribute__((ext_vector_type(16))) __bf16        v16bf;
typedef __attribute__((ext_vector_type(8)))  float         v8f;
typedef __attribute__((ext_vector_type(4)))  unsigned int  u32x4;
typedef __attribute__((ext_vector_type(4)))  int           i32x4;

typedef __attribute__((address_space(1))) i32x4* gvec_t;   // global int32x4*
typedef __attribute__((address_space(3))) i32x4* lvec_t;   // LDS    int32x4*

// 32-byte container for one WMMA 16-bit A/B operand (16 bf16 per lane)
union ABop {
    struct { u32x4 lo, hi; } q;
    v16bf v;
};

#define WMMA_BF16(A, Bm, C) \
    __builtin_amdgcn_wmma_f32_16x16x32_bf16(false, (A), false, (Bm), (short)0, (C), false, false)

// async Global -> LDS copy of 16 bytes per lane (ASYNCcnt-tracked)
#define ASYNC_CP16(gsrc, ldst) \
    __builtin_amdgcn_global_load_async_to_lds_b128((gvec_t)(gsrc), (lvec_t)(ldst), 0, 0)

// ---------------------------------------------------------------------------
// Kernel 1: transpose+convert  x [B,D,L] f32  ->  xT [B,L,D] bf16
// ---------------------------------------------------------------------------
__global__ __launch_bounds__(256) void convx_kernel(const float* __restrict__ x,
                                                    __bf16* __restrict__ xT)
{
    __shared__ float t[32][33];
    const int b  = blockIdx.z;
    const int d0 = blockIdx.y * 32;
    const int l0 = blockIdx.x * 32;
    const int tx = threadIdx.x, ty = threadIdx.y;
#pragma unroll
    for (int j = 0; j < 4; ++j)
        t[ty + 8 * j][tx] = x[((size_t)b * DD + d0 + ty + 8 * j) * LL + l0 + tx];
    __syncthreads();
#pragma unroll
    for (int j = 0; j < 4; ++j)
        xT[((size_t)b * LL + l0 + ty + 8 * j) * DD + d0 + tx] = (__bf16)t[tx][ty + 8 * j];
}

// ---------------------------------------------------------------------------
// Kernel 2: Wc[0:1024)   = Wq (bf16),  Wc[1024:3072) = Wkv (bf16)
// ---------------------------------------------------------------------------
__global__ __launch_bounds__(256) void convw_kernel(const float* __restrict__ Wq,
                                                    const float* __restrict__ Wkv,
                                                    __bf16* __restrict__ Wc)
{
    const size_t i = (size_t)blockIdx.x * 256 + threadIdx.x;   // < 3072*1024
    const size_t nq = (size_t)DD * DD;
    float v = (i < nq) ? Wq[i] : Wkv[i - nq];
    Wc[i] = (__bf16)v;
}

// ---------------------------------------------------------------------------
// Tile store helper: route a 16x16 C tile (row0 in [0,8192), o0 in [0,3072))
// to Q (scaled, [B,H,L,dh]), K ([B,H,L,dh]) or V-transposed ([B,H,dh,L]).
// ---------------------------------------------------------------------------
__device__ __forceinline__ void store_tile(v8f c, int row0, int o0, int lane,
                                           __bf16* __restrict__ Qw,
                                           __bf16* __restrict__ Kw,
                                           __bf16* __restrict__ Vtw)
{
    const int n     = lane & 15;
    const int half8 = (lane < 16) ? 0 : 8;     // C-layout row offset
    const int bI    = row0 >> 11;              // batch (L = 2048)
    const int l0    = row0 & (LL - 1);

    if (o0 < DD) {                             // ---- Q (scaled by dh^-0.5) ----
        const int h = o0 >> 6, dd = (o0 & 63) + n;
        __bf16* dst = Qw + ((size_t)(bI * HH + h) * LL + l0 + half8) * DH + dd;
#pragma unroll
        for (int r = 0; r < 8; ++r) dst[(size_t)r * DH] = (__bf16)(c[r] * 0.125f);
    } else if (o0 < 2 * DD) {                  // ---- K ----
        const int oo = o0 - DD;
        const int h = oo >> 6, dd = (oo & 63) + n;
        __bf16* dst = Kw + ((size_t)(bI * HH + h) * LL + l0 + half8) * DH + dd;
#pragma unroll
        for (int r = 0; r < 8; ++r) dst[(size_t)r * DH] = (__bf16)c[r];
    } else {                                   // ---- V (transposed) ----
        const int oo = o0 - 2 * DD;
        const int h = oo >> 6, dd = (oo & 63) + n;
        __bf16* dst = Vtw + ((size_t)(bI * HH + h) * DH + dd) * LL + l0 + half8;
#pragma unroll
        for (int r = 0; r < 8; ++r) dst[r] = (__bf16)c[r];
    }
}

// ---------------------------------------------------------------------------
// Kernel 3: LDS-blocked projection GEMM with async double-buffered staging.
//   out[row,o] = sum_d xT[row,d] * Wc[o,d],  rows 8192, cols 3072.
//   Block (256 thr / 8 waves) -> 128x128 tile; per 32-K step async-stage
//   A(128x32) + B(128x32) bf16 into LDS; wave (mw,nw) computes 32x64 -> 8 WMMA.
// ---------------------------------------------------------------------------
#define PBM 128
#define PBN 128
#define PBK 32

__global__ __launch_bounds__(256) void proj_kernel(const __bf16* __restrict__ xT,
                                                   const __bf16* __restrict__ Wc,
                                                   __bf16* __restrict__ Qw,
                                                   __bf16* __restrict__ Kw,
                                                   __bf16* __restrict__ Vtw)
{
    __shared__ __align__(16) __bf16 lsA[2][PBM * PBK];   // 2 x 8 KB
    __shared__ __align__(16) __bf16 lsB[2][PBN * PBK];   // 2 x 8 KB

    const int tid  = threadIdx.x;
    const int lane = tid & 31;
    const int wv   = tid >> 5;            // 0..7
    const int mw   = wv & 3;              // 32-row slab within 128
    const int nw   = wv >> 2;             // 64-col slab within 128
    const int nbt  = (3 * DD) / PBN;      // 24 column blocks
    const int rowA0 = (blockIdx.x / nbt) * PBM;
    const int col0  = (blockIdx.x % nbt) * PBN;
    const int n    = lane & 15;
    const int koff = (lane < 16) ? 0 : 8; // ISA 16-bit A/B lane striping

    // per-thread staging: 2 chunks of A + 2 chunks of B (16 B each) = 4 asyncs
    auto stage = [&](int buf, int kk) {
#pragma unroll
        for (int rdi = 0; rdi < 2; ++rdi) {
            const int cidx = rdi * 256 + tid;
            const int row  = cidx >> 2;
            const int ch   = (cidx & 3) * 8;
            ASYNC_CP16(&xT[(size_t)(rowA0 + row) * DD + kk + ch],
                       &lsA[buf][row * PBK + ch]);
            ASYNC_CP16(&Wc[(size_t)(col0 + row) * DD + kk + ch],
                       &lsB[buf][row * PBK + ch]);
        }
    };

    v8f c[2][4] = {};
    stage(0, 0);

    for (int kk = 0; kk < DD; kk += PBK) {
        const int cur = (kk >> 5) & 1;
        if (kk + PBK < DD) {
            stage(cur ^ 1, kk + PBK);                         // overlap next tile
            asm volatile("s_wait_asynccnt 0x4" ::: "memory"); // current tile done
        } else {
            asm volatile("s_wait_asynccnt 0x0" ::: "memory");
        }
        __syncthreads();

        ABop a[2], bo[4];
#pragma unroll
        for (int mt = 0; mt < 2; ++mt) {
            const __bf16* p = &lsA[cur][(mw * 32 + mt * 16 + n) * PBK];
            a[mt].q.lo = *(const u32x4*)(p + koff);
            a[mt].q.hi = *(const u32x4*)(p + koff + 16);
        }
#pragma unroll
        for (int nt = 0; nt < 4; ++nt) {
            const __bf16* p = &lsB[cur][(nw * 64 + nt * 16 + n) * PBK];
            bo[nt].q.lo = *(const u32x4*)(p + koff);
            bo[nt].q.hi = *(const u32x4*)(p + koff + 16);
        }
#pragma unroll
        for (int mt = 0; mt < 2; ++mt)
#pragma unroll
            for (int nt = 0; nt < 4; ++nt)
                c[mt][nt] = WMMA_BF16(a[mt].v, bo[nt].v, c[mt][nt]);
        __syncthreads();   // all waves done reading buf before it is re-filled
    }

#pragma unroll
    for (int mt = 0; mt < 2; ++mt)
#pragma unroll
        for (int nt = 0; nt < 4; ++nt)
            store_tile(c[mt][nt], rowA0 + mw * 32 + mt * 16,
                       col0 + nw * 64 + nt * 16, lane, Qw, Kw, Vtw);
}

// ---------------------------------------------------------------------------
// Kernel 4: flash attention. Block = 4 waves, all same (b,h); each wave owns
// 16 queries. Per 64-key block: K/V async-staged (double-buffered) in LDS for
// all 4 waves, S = Q K^T (8 wmma), mask, online softmax, P via per-wave LDS,
// O += P V (8 wmma).
// ---------------------------------------------------------------------------
__global__ __launch_bounds__(128) void attn_kernel(const __bf16* __restrict__ Q,
                                                   const __bf16* __restrict__ Kw,
                                                   const __bf16* __restrict__ Vt,
                                                   const int* __restrict__ mask,
                                                   float* __restrict__ out)
{
    __shared__ __align__(16) __bf16 lsK[2][64 * 64];     // [token][dh]  2 x 8 KB
    __shared__ __align__(16) __bf16 lsV[2][64 * 64];     // [dh][token]  2 x 8 KB
    __shared__ __align__(16) __bf16 plds[4][16 * 64];    // P staging    8 KB

    const int tid  = threadIdx.x;
    const int lane = tid & 31;
    const int wv   = tid >> 5;
    const int bh   = blockIdx.x >> 5;                    // 32 blocks per (b,h)
    const int qt   = (blockIdx.x & 31) * 4 + wv;         // query tile 0..127
    const int b    = bh >> 4;
    const int h    = bh & 15;

    const __bf16* Qb = Q  + ((size_t)bh * LL + qt * 16) * DH;
    const __bf16* Kb = Kw + (size_t)bh * LL * DH;
    const __bf16* Vb = Vt + (size_t)bh * DH * LL;
    const int*    mb = mask + b * LL;

    const int n     = lane & 15;
    const int koff  = (lane < 16) ? 0 : 8;
    const int half8 = (lane < 16) ? 0 : 8;

    // per-thread staging: 4 chunks of K + 4 chunks of V (16 B each) = 8 asyncs
    auto stageKV = [&](int buf, int t0) {
#pragma unroll
        for (int rdi = 0; rdi < 4; ++rdi) {
            const int cidx = rdi * 128 + tid;   // 512 chunks of 16 B each
            const int row  = cidx >> 3;
            const int ch   = (cidx & 7) * 8;
            ASYNC_CP16(&Kb[(size_t)(t0 + row) * DH + ch], &lsK[buf][row * 64 + ch]);
            ASYNC_CP16(&Vb[(size_t)row * LL + t0 + ch],   &lsV[buf][row * 64 + ch]);
        }
    };

    // 16x64 Q tile as two 16x32 A operands, resident for the whole sweep.
    ABop a0, a1;
    {
        const __bf16* qrow = Qb + n * DH;
        a0.q.lo = *(const u32x4*)(qrow + koff);
        a0.q.hi = *(const u32x4*)(qrow + koff + 16);
        a1.q.lo = *(const u32x4*)(qrow + 32 + koff);
        a1.q.hi = *(const u32x4*)(qrow + 32 + koff + 16);
    }

    v8f oacc[4] = {v8f{}, v8f{}, v8f{}, v8f{}};
    float mrow[8], srow[8];
#pragma unroll
    for (int r = 0; r < 8; ++r) { mrow[r] = -__builtin_inff(); srow[r] = 0.0f; }

    __bf16* pl = &plds[wv][0];
    stageKV(0, 0);

    for (int t0 = 0; t0 < LL; t0 += 64) {
        const int cur = (t0 >> 6) & 1;
        if (t0 + 64 < LL) {
            stageKV(cur ^ 1, t0 + 64);                        // overlap next block
            asm volatile("s_wait_asynccnt 0x8" ::: "memory"); // current block done
        } else {
            asm volatile("s_wait_asynccnt 0x0" ::: "memory");
        }
        __syncthreads();

        // ---- S = Q K^T over 64 keys (four 16-key column groups) ----
        v8f s[4];
#pragma unroll
        for (int kg = 0; kg < 4; ++kg) {
            const __bf16* kp = &lsK[cur][(kg * 16 + n) * 64];
            ABop bk;
            bk.q.lo = *(const u32x4*)(kp + koff);
            bk.q.hi = *(const u32x4*)(kp + koff + 16);
            v8f t = {};
            t = WMMA_BF16(a0.v, bk.v, t);
            bk.q.lo = *(const u32x4*)(kp + 32 + koff);
            bk.q.hi = *(const u32x4*)(kp + 32 + koff + 16);
            t = WMMA_BF16(a1.v, bk.v, t);
            s[kg] = t;
        }

        int mk[4];
#pragma unroll
        for (int kg = 0; kg < 4; ++kg) mk[kg] = mb[t0 + kg * 16 + n];

        // ---- mask + online softmax (row stats follow C-layout striping) ----
#pragma unroll
        for (int r = 0; r < 8; ++r) {
            float x[4];
#pragma unroll
            for (int kg = 0; kg < 4; ++kg) x[kg] = mk[kg] ? s[kg][r] : NEG_INF;
            float mx = fmaxf(fmaxf(x[0], x[1]), fmaxf(x[2], x[3]));
            mx = fmaxf(mx, __shfl_xor(mx, 1));
            mx = fmaxf(mx, __shfl_xor(mx, 2));
            mx = fmaxf(mx, __shfl_xor(mx, 4));
            mx = fmaxf(mx, __shfl_xor(mx, 8));
            const float mn = fmaxf(mrow[r], mx);
            const float al = __expf(mrow[r] - mn);
            float p[4], ps = 0.0f;
#pragma unroll
            for (int kg = 0; kg < 4; ++kg) { p[kg] = __expf(x[kg] - mn); ps += p[kg]; }
            ps += __shfl_xor(ps, 1);
            ps += __shfl_xor(ps, 2);
            ps += __shfl_xor(ps, 4);
            ps += __shfl_xor(ps, 8);
            srow[r] = srow[r] * al + ps;
            mrow[r] = mn;
            const int row = r + half8;
#pragma unroll
            for (int kg = 0; kg < 4; ++kg)
                pl[row * 64 + kg * 16 + n] = (__bf16)p[kg];
            oacc[0][r] *= al; oacc[1][r] *= al; oacc[2][r] *= al; oacc[3][r] *= al;
        }
        asm volatile("s_wait_dscnt 0" ::: "memory");

        // ---- reload P as two 16x32 A operands ----
        ABop pa0, pa1;
        {
            const __bf16* pr = pl + n * 64;
            pa0.q.lo = *(const u32x4*)(pr + koff);
            pa0.q.hi = *(const u32x4*)(pr + koff + 16);
            pa1.q.lo = *(const u32x4*)(pr + 32 + koff);
            pa1.q.hi = *(const u32x4*)(pr + 32 + koff + 16);
        }

        // ---- O += P * V (V^T in LDS: row = dh, contiguous tokens) ----
#pragma unroll
        for (int g = 0; g < 4; ++g) {
            const __bf16* vp = &lsV[cur][(g * 16 + n) * 64];
            ABop bv;
            bv.q.lo = *(const u32x4*)(vp + koff);
            bv.q.hi = *(const u32x4*)(vp + koff + 16);
            oacc[g] = WMMA_BF16(pa0.v, bv.v, oacc[g]);
            bv.q.lo = *(const u32x4*)(vp + 32 + koff);
            bv.q.hi = *(const u32x4*)(vp + 32 + koff + 16);
            oacc[g] = WMMA_BF16(pa1.v, bv.v, oacc[g]);
        }
        __syncthreads();   // all waves done reading buf before it is re-filled
    }

    // ---- normalize and store out[B, D, L] f32 ----
    float inv[8];
#pragma unroll
    for (int r = 0; r < 8; ++r) inv[r] = 1.0f / srow[r];
#pragma unroll
    for (int g = 0; g < 4; ++g) {
        const int ch = h * DH + g * 16 + n;
        float* ob = out + ((size_t)b * DD + ch) * LL + qt * 16 + half8;
#pragma unroll
        for (int r = 0; r < 8; ++r) ob[r] = oacc[g][r] * inv[r];
    }
}

// ---------------------------------------------------------------------------
// Host-side launch. Workspace layout (bytes):
//   xT : 0          (16 MB)   Wc : 16777216 (6 MB)
//   Q  : 23068672   (16 MB)   K  : 39845888 (16 MB)   Vt : 56623104 (16 MB)
// ---------------------------------------------------------------------------
extern "C" void kernel_launch(void* const* d_in, const int* in_sizes, int n_in,
                              void* d_out, int out_size, void* d_ws, size_t ws_size,
                              hipStream_t stream) {
    const float* x    = (const float*)d_in[0];
    const float* Wq   = (const float*)d_in[1];
    const float* Wkv  = (const float*)d_in[2];
    const int*   mask = (const int*)d_in[3];
    float*       out  = (float*)d_out;

    char* ws = (char*)d_ws;
    __bf16* xT  = (__bf16*)(ws);
    __bf16* Wc  = (__bf16*)(ws + 16777216);
    __bf16* Qw  = (__bf16*)(ws + 23068672);
    __bf16* Kw  = (__bf16*)(ws + 39845888);
    __bf16* Vtw = (__bf16*)(ws + 56623104);

    convx_kernel<<<dim3(LL / 32, DD / 32, BB), dim3(32, 8), 0, stream>>>(x, xT);
    convw_kernel<<<(3 * DD * DD) / 256, 256, 0, stream>>>(Wq, Wkv, Wc);
    proj_kernel<<<(8192 / PBM) * ((3 * DD) / PBN), 256, 0, stream>>>(xT, Wc, Qw, Kw, Vtw);
    attn_kernel<<<(BB * HH * (LL / 16)) / 4, 128, 0, stream>>>(Qw, Kw, Vtw, mask, out);
}
